// KSimplexLinear_6047313953380
// MI455X (gfx1250) — compile-verified, matrix operand order checked by reading, not run on hardware
//
#include <hip/hip_runtime.h>

#define NB 1024
#define ND 2048
#define HH 5
#define EE 4
#define DEPTH 5

typedef __attribute__((ext_vector_type(16))) __bf16        v16bf;
typedef __attribute__((ext_vector_type(8)))  float         v8f;
typedef __attribute__((ext_vector_type(8)))  unsigned int  v8u;

static __device__ __forceinline__ unsigned short f2bf(float f) {
    union { float f; unsigned u; } x; x.f = f;
    unsigned r = x.u + 0x7fffu + ((x.u >> 16) & 1u);   // round-to-nearest-even
    return (unsigned short)(r >> 16);
}
static __device__ __forceinline__ unsigned pkbf(float lo, float hi) {
    return (unsigned)f2bf(lo) | ((unsigned)f2bf(hi) << 16);
}
static __device__ __forceinline__ v16bf mk_mat(unsigned u0, unsigned u1, unsigned u2) {
    v8u t = {u0, u1, u2, 0u, 0u, 0u, 0u, 0u};
    return __builtin_bit_cast(v16bf, t);
}
static __device__ __forceinline__ float gelu_exact(float x) {
    return 0.5f * x * (1.0f + erff(x * 0.70710678118654752f));
}
static __device__ __forceinline__ float fast_rcp(float x) {
#if __has_builtin(__builtin_amdgcn_rcpf)
    return __builtin_amdgcn_rcpf(x);
#else
    return 1.0f / x;
#endif
}

__global__ __launch_bounds__(256) void ksimplex_wmma(
    const float* __restrict__ x,
    const float* __restrict__ entry_w,   // [H,1]
    const float* __restrict__ entry_b,   // [H]
    const float* __restrict__ route_w,   // [DEPTH,E,H]
    const float* __restrict__ route_b,   // [DEPTH,E]
    const float* __restrict__ edge_w,    // [DEPTH,E,H,H]
    const float* __restrict__ layer_bias,// [DEPTH,H]
    const float* __restrict__ exit_w,    // [1,H]
    const float* __restrict__ exit_b,    // [1]
    float* __restrict__ out)
{
    const int lane  = threadIdx.x & 31;
    const int wid   = threadIdx.x >> 5;
    const int lo    = lane & 15;
    const bool isLow = (lane < 16);
    const long base = ((long)blockIdx.x * 8 + wid) * 16;   // 16 elements per wave

    // ---- Hoist all layer A-tiles (bf16 WMMA layout) into registers.
    // Combined rows j: 0..19 edge_w[i][j/5][j%5][:], 20..23 route rows (+bias at K=5),
    // 24..27 DUPLICATED route rows (so high lanes of the D layout see the logits too).
    unsigned a0u[DEPTH][3], a1u[DEPTH][3];
    #pragma unroll
    for (int i = 0; i < DEPTH; ++i) {
        #pragma unroll
        for (int t = 0; t < 2; ++t) {
            unsigned u0 = 0u, u1 = 0u, u2 = 0u;
            if (isLow) {
                const int j = t * 16 + lane;
                int rrow = -1;
                const float* p = nullptr;
                if (j < 20)       p = edge_w + (((i * EE) + (j / 5)) * HH + (j % 5)) * HH;
                else if (j < 24)  rrow = j - 20;
                else if (j < 28)  rrow = j - 24;   // duplicate routing rows for high lanes
                float bias = 0.f;
                if (rrow >= 0) {
                    p    = route_w + ((i * EE) + rrow) * HH;
                    bias = route_b[i * EE + rrow];
                }
                if (p) {
                    float w[5];
                    #pragma unroll
                    for (int k = 0; k < 5; ++k) w[k] = p[k];
                    u0 = pkbf(w[0], w[1]);
                    u1 = pkbf(w[2], w[3]);
                    u2 = pkbf(w[4], bias);   // K=5 column multiplies B's constant 1.0
                }
            }
            if (t == 0) { a0u[i][0] = u0; a0u[i][1] = u1; a0u[i][2] = u2; }
            else        { a1u[i][0] = u0; a1u[i][1] = u1; a1u[i][2] = u2; }
        }
    }

    // ---- Entry: h = x*entry_w + entry_b (lane pair n / n+16 replicates element n)
    const float xv = x[base + lo];
    float h[HH];
    #pragma unroll
    for (int k = 0; k < HH; ++k) h[k] = fmaf(xv, entry_w[k], entry_b[k]);

    const v8f czero = {0.f, 0.f, 0.f, 0.f, 0.f, 0.f, 0.f, 0.f};

    #pragma unroll
    for (int i = 0; i < DEPTH; ++i) {
        // ---- B: element n in column n (lanes 0-15); K=0..4 = h, K=5 = 1.0 (bias lane)
        unsigned b0 = 0u, b1 = 0u, b2 = 0u;
        if (isLow) {
            b0 = pkbf(h[0], h[1]);
            b1 = pkbf(h[2], h[3]);
            b2 = pkbf(h[4], 1.0f);
        }
        const v16bf Bm = mk_mat(b0, b1, b2);
        const v16bf A0 = mk_mat(a0u[i][0], a0u[i][1], a0u[i][2]);
        const v16bf A1 = mk_mat(a1u[i][0], a1u[i][1], a1u[i][2]);

        // D[j][n] ; rows 0-15 in d0, rows 16-27 in d1 (24-27 = route duplicates)
        v8f d0 = __builtin_amdgcn_wmma_f32_16x16x32_bf16(
            false, A0, false, Bm, (short)0, czero, false, false);
        v8f d1 = __builtin_amdgcn_wmma_f32_16x16x32_bf16(
            false, A1, false, Bm, (short)0, czero, false, false);

        // ---- softmax: low lanes read logits from d1[4..7], high lanes from d1[0..3]
        float l0 = isLow ? d1[4] : d1[0];
        float l1 = isLow ? d1[5] : d1[1];
        float l2 = isLow ? d1[6] : d1[2];
        float l3 = isLow ? d1[7] : d1[3];
        float m  = fmaxf(fmaxf(l0, l1), fmaxf(l2, l3));
        float e0 = __expf(l0 - m), e1 = __expf(l1 - m);
        float e2 = __expf(l2 - m), e3 = __expf(l3 - m);
        float inv = fast_rcp(e0 + e1 + e2 + e3);
        float rw0 = e0 * inv, rw1 = e1 * inv, rw2 = e2 * inv, rw3 = e3 * inv;

        // ---- mixture: each half accumulates the eo rows it holds
        float p0, p1, p2, p3, p4;
        if (isLow) {
            // low lane holds j=0..7 (d0) and j=16..19 (d1[0..3])
            p0 = fmaf(rw1, d0[5], rw0 * d0[0]);
            p1 = fmaf(rw3, d1[0], fmaf(rw1, d0[6], rw0 * d0[1]));
            p2 = fmaf(rw3, d1[1], fmaf(rw1, d0[7], rw0 * d0[2]));
            p3 = fmaf(rw3, d1[2], rw0 * d0[3]);
            p4 = fmaf(rw3, d1[3], rw0 * d0[4]);
        } else {
            // high lane holds j=8..15 in d0[0..7]
            p0 = fmaf(rw3, d0[7], rw2 * d0[2]);
            p1 = rw2 * d0[3];
            p2 = rw2 * d0[4];
            p3 = fmaf(rw2, d0[5], rw1 * d0[0]);
            p4 = fmaf(rw2, d0[6], rw1 * d0[1]);
        }
        float p[HH] = {p0, p1, p2, p3, p4};
        #pragma unroll
        for (int o = 0; o < HH; ++o) {
            float q = p[o] + __shfl_xor(p[o], 16, 32);   // merge lane-pair partials
            h[o] = gelu_exact(q + layer_bias[i * HH + o]);
        }
    }

    // ---- exit projection
    float acc = exit_b[0];
    #pragma unroll
    for (int o = 0; o < HH; ++o) acc = fmaf(h[o], exit_w[o], acc);
    if (isLow) out[base + lo] = acc;
}

extern "C" void kernel_launch(void* const* d_in, const int* in_sizes, int n_in,
                              void* d_out, int out_size, void* d_ws, size_t ws_size,
                              hipStream_t stream) {
    (void)in_sizes; (void)n_in; (void)out_size; (void)d_ws; (void)ws_size;
    const float* x          = (const float*)d_in[0];
    const float* entry_w    = (const float*)d_in[1];
    const float* entry_b    = (const float*)d_in[2];
    const float* route_w    = (const float*)d_in[3];
    const float* route_b    = (const float*)d_in[4];
    const float* edge_w     = (const float*)d_in[5];
    const float* layer_bias = (const float*)d_in[6];
    const float* exit_w     = (const float*)d_in[7];
    const float* exit_b     = (const float*)d_in[8];
    float* out = (float*)d_out;

    const int total   = NB * ND;          // 2,097,152 elements
    const int perBlk  = 8 * 16;           // 8 waves * 16 elements
    const int nblocks = total / perBlk;   // 16384
    ksimplex_wmma<<<nblocks, 256, 0, stream>>>(
        x, entry_w, entry_b, route_w, route_b, edge_w, layer_bias,
        exit_w, exit_b, out);
}
